// SelectiveMemoryTransformer_57028575756938
// MI455X (gfx1250) — matrix operand order repaired and checked by reading, not compiled, not run on hardware
//
#include <hip/hip_runtime.h>

// ---------------------------------------------------------------------------
// SelectiveMemoryTransformer for MI455X (gfx1250), wave32 + WMMA bf16.
//   B=4 T=64 L=128 D=768 H=12 dh=64 F=3072 S=8 K=5  -> 32768 tokens.
// All dense GEMMs use v_wmma_f32_16x16x32_bf16 (f32 accum). Attention is a
// fused per-(seq,head) kernel: QK^T (WMMA) -> masked softmax in registers
// (wave32 shfl_xor over 16-lane halves) -> P*V (WMMA). GRU scan is the only
// sequential part and runs as a 1024-thread LDS-resident kernel per batch.
// ---------------------------------------------------------------------------

typedef __attribute__((ext_vector_type(16))) __bf16 v16bf;
typedef __attribute__((ext_vector_type(8)))  float  v8f;
typedef __attribute__((ext_vector_type(4)))  unsigned int u32x4;

union Frag { v16bf v; u32x4 q[2]; };

static __device__ __forceinline__ unsigned short f2bf(float f) {
  union { float f; unsigned u; } c; c.f = f;
  unsigned r = c.u + 0x7FFFu + ((c.u >> 16) & 1u);   // round-to-nearest-even
  return (unsigned short)(r >> 16);
}
static __device__ __forceinline__ float sigmoidf_(float x) { return 1.f / (1.f + expf(-x)); }
static __device__ __forceinline__ v8f vzero() {
  v8f z;
#pragma unroll
  for (int e = 0; e < 8; ++e) z[e] = 0.f;
  return z;
}
static __device__ __forceinline__ v8f wmma_bf(v16bf a, v16bf b, v8f c) {
  return __builtin_amdgcn_wmma_f32_16x16x32_bf16(false, a, false, b, (short)0, c, false, false);
}

// ------------------------------ f32 -> bf16 --------------------------------
__global__ __launch_bounds__(256) void cvt_k(const float* __restrict__ in,
                                             unsigned short* __restrict__ out, long n4) {
  long i = (long)blockIdx.x * 256 + threadIdx.x;
  if (i < n4) {
    float4 v = *(const float4*)&in[i * 4];
    out[i * 4 + 0] = f2bf(v.x); out[i * 4 + 1] = f2bf(v.y);
    out[i * 4 + 2] = f2bf(v.z); out[i * 4 + 3] = f2bf(v.w);
  }
}

// ------------------------------- GEMM --------------------------------------
// C[M,N] = act(A[M,K] * W[N,K]^T + bias).  W is bf16. A is f32 (ABF=0) or
// bf16 (ABF=1). Output f32 (OBF=0) or bf16 (OBF=1). ACT: 0=none, 1=exact GELU.
// Block = 128x128 tile, 8 waves (2x4), each wave = 64x32 = 4x2 WMMA tiles.
template <int ACT, int ABF, int OBF>
__global__ __launch_bounds__(256) void gemm_k(const void* __restrict__ Av,
                                              const unsigned short* __restrict__ Wb,
                                              const float* __restrict__ bias,
                                              void* __restrict__ Cv,
                                              int M, int N, int Kd) {
  __shared__ unsigned short sA[128 * 32];
  __shared__ unsigned short sB[128 * 32];
  const int tid = threadIdx.x;
  const int lane = tid & 31, wid = tid >> 5;
  const int wm = wid & 1, wn = wid >> 1;     // 2 x 4 wave grid
  const int r = lane & 15, hl = lane >> 4;
  const long rowBase = (long)blockIdx.y * 128;
  const long colBase = (long)blockIdx.x * 128;
  const float* Af = (const float*)Av;
  const unsigned short* Ab = (const unsigned short*)Av;

  v8f acc[4][2];
#pragma unroll
  for (int mt = 0; mt < 4; ++mt)
#pragma unroll
    for (int nt = 0; nt < 2; ++nt) acc[mt][nt] = vzero();

  for (int kt = 0; kt < Kd; kt += 32) {
    if (ABF) {
#pragma unroll
      for (int i = 0; i < 2; ++i) {                 // 512 chunks of 8 halfs
        int idx = tid + i * 256;
        int row = idx >> 2, c8 = (idx & 3) * 8;
        *(u32x4*)&sA[row * 32 + c8] =
            *(const u32x4*)&Ab[(rowBase + row) * Kd + kt + c8];
      }
    } else {
#pragma unroll
      for (int i = 0; i < 4; ++i) {                 // 1024 chunks of 4 floats
        int idx = tid + i * 256;
        int row = idx >> 3, c4 = (idx & 7) * 4;
        float4 v = *(const float4*)&Af[(rowBase + row) * Kd + kt + c4];
        unsigned u0 = (unsigned)f2bf(v.x) | ((unsigned)f2bf(v.y) << 16);
        unsigned u1 = (unsigned)f2bf(v.z) | ((unsigned)f2bf(v.w) << 16);
        *(unsigned*)&sA[row * 32 + c4]     = u0;
        *(unsigned*)&sA[row * 32 + c4 + 2] = u1;
      }
    }
#pragma unroll
    for (int i = 0; i < 2; ++i) {
      int idx = tid + i * 256;
      int row = idx >> 2, c8 = (idx & 3) * 8;
      *(u32x4*)&sB[row * 32 + c8] =
          *(const u32x4*)&Wb[(colBase + row) * Kd + kt + c8];
    }
    __syncthreads();

    Frag a[4], b[2];
#pragma unroll
    for (int mt = 0; mt < 4; ++mt) {
      int arow = wm * 64 + mt * 16 + r;
      a[mt].q[0] = *(const u32x4*)&sA[arow * 32 + hl * 8];
      a[mt].q[1] = *(const u32x4*)&sA[arow * 32 + 16 + hl * 8];
    }
#pragma unroll
    for (int nt = 0; nt < 2; ++nt) {
      int brow = wn * 32 + nt * 16 + r;
      b[nt].q[0] = *(const u32x4*)&sB[brow * 32 + hl * 8];
      b[nt].q[1] = *(const u32x4*)&sB[brow * 32 + 16 + hl * 8];
    }
#pragma unroll
    for (int mt = 0; mt < 4; ++mt)
#pragma unroll
      for (int nt = 0; nt < 2; ++nt)
        acc[mt][nt] = wmma_bf(a[mt].v, b[nt].v, acc[mt][nt]);
    __syncthreads();
  }

  float* Cf = (float*)Cv;
  unsigned short* Cb = (unsigned short*)Cv;
#pragma unroll
  for (int mt = 0; mt < 4; ++mt)
#pragma unroll
    for (int nt = 0; nt < 2; ++nt)
#pragma unroll
      for (int v = 0; v < 8; ++v) {
        long grow = rowBase + wm * 64 + mt * 16 + v + 8 * hl;
        long gcol = colBase + wn * 32 + nt * 16 + r;
        float val = acc[mt][nt][v] + bias[gcol];
        if (ACT == 1) val = 0.5f * val * (1.f + erff(val * 0.70710678118f));
        if (OBF) Cb[grow * N + gcol] = f2bf(val);
        else     Cf[grow * N + gcol] = val;
      }
}

// ---------------------------- fused attention ------------------------------
// One block per (seq n, head h). Q/K/V bf16 [n,l,D] head slices. dh=64, L=128.
__global__ __launch_bounds__(256) void attn_k(const unsigned short* __restrict__ Q,
                                              const unsigned short* __restrict__ K,
                                              const unsigned short* __restrict__ V,
                                              const int* __restrict__ mask,
                                              float* __restrict__ O, int H) {
  __shared__ unsigned short sm[24576];                 // 48 KB
  unsigned short* Ps = sm;                             // [128][128], reuses Q/K
  unsigned short* Qs = sm;                             // [128][64]
  unsigned short* Ks = sm + 8192;                      // [128][64]
  unsigned short* Vt = sm + 16384;                     // [64][128] (transposed)
  const int tid = threadIdx.x, lane = tid & 31, wid = tid >> 5;
  const int r = lane & 15, hl = lane >> 4;
  const int n = blockIdx.x / H, h = blockIdx.x % H;
  const long qb = ((long)n * 128) * 768 + h * 64;

#pragma unroll
  for (int i = 0; i < 4; ++i) {
    int idx = tid + i * 256;
    int l = idx >> 3, c8 = (idx & 7) * 8;
    *(u32x4*)&Qs[l * 64 + c8] = *(const u32x4*)&Q[qb + (long)l * 768 + c8];
    *(u32x4*)&Ks[l * 64 + c8] = *(const u32x4*)&K[qb + (long)l * 768 + c8];
  }
#pragma unroll
  for (int i = 0; i < 4; ++i) {
    int idx = tid + i * 256;
    int l = idx >> 3, c8 = (idx & 7) * 8;
    u32x4 pk = *(const u32x4*)&V[qb + (long)l * 768 + c8];
    const unsigned short* up = (const unsigned short*)&pk;
#pragma unroll
    for (int j = 0; j < 8; ++j) Vt[(c8 + j) * 128 + l] = up[j];
  }
  __syncthreads();

  const int* am = mask + (long)n * 128;
  float mb[8];
#pragma unroll
  for (int j = 0; j < 8; ++j) mb[j] = (am[j * 16 + r] > 0) ? 0.f : -1e30f;

  v8f s[8];
#pragma unroll
  for (int j = 0; j < 8; ++j) s[j] = vzero();

#pragma unroll
  for (int kc = 0; kc < 2; ++kc) {
    Frag aq;
    int arow = wid * 16 + r;
    aq.q[0] = *(const u32x4*)&Qs[arow * 64 + kc * 32 + hl * 8];
    aq.q[1] = *(const u32x4*)&Qs[arow * 64 + kc * 32 + 16 + hl * 8];
#pragma unroll
    for (int j = 0; j < 8; ++j) {
      Frag bk;
      int brow = j * 16 + r;
      bk.q[0] = *(const u32x4*)&Ks[brow * 64 + kc * 32 + hl * 8];
      bk.q[1] = *(const u32x4*)&Ks[brow * 64 + kc * 32 + 16 + hl * 8];
      s[j] = wmma_bf(aq.v, bk.v, s[j]);
    }
  }
#pragma unroll
  for (int j = 0; j < 8; ++j)
#pragma unroll
    for (int v = 0; v < 8; ++v) s[j][v] = s[j][v] * 0.125f + mb[j];

  // softmax: row = (v, lane-half); 128 entries live in 16 lanes x 8 j-tiles
#pragma unroll
  for (int v = 0; v < 8; ++v) {
    float mx = -3.4e38f;
#pragma unroll
    for (int j = 0; j < 8; ++j) mx = fmaxf(mx, s[j][v]);
    for (int o = 1; o < 16; o <<= 1) mx = fmaxf(mx, __shfl_xor(mx, o, 32));
    float sum = 0.f;
#pragma unroll
    for (int j = 0; j < 8; ++j) { float e = expf(s[j][v] - mx); s[j][v] = e; sum += e; }
    for (int o = 1; o < 16; o <<= 1) sum += __shfl_xor(sum, o, 32);
    float inv = 1.f / sum;
#pragma unroll
    for (int j = 0; j < 8; ++j) s[j][v] *= inv;
  }
  __syncthreads();                                   // done reading Qs/Ks
#pragma unroll
  for (int j = 0; j < 8; ++j)
#pragma unroll
    for (int v = 0; v < 8; ++v) {
      int row = wid * 16 + v + 8 * hl;
      Ps[row * 128 + j * 16 + r] = f2bf(s[j][v]);
    }
  __syncthreads();

  v8f o[4];
#pragma unroll
  for (int nt = 0; nt < 4; ++nt) o[nt] = vzero();
#pragma unroll
  for (int kc = 0; kc < 4; ++kc) {
    Frag ap;
    int arow = wid * 16 + r;
    ap.q[0] = *(const u32x4*)&Ps[arow * 128 + kc * 32 + hl * 8];
    ap.q[1] = *(const u32x4*)&Ps[arow * 128 + kc * 32 + 16 + hl * 8];
#pragma unroll
    for (int nt = 0; nt < 4; ++nt) {
      Frag bv;
      int d = nt * 16 + r;
      bv.q[0] = *(const u32x4*)&Vt[d * 128 + kc * 32 + hl * 8];
      bv.q[1] = *(const u32x4*)&Vt[d * 128 + kc * 32 + 16 + hl * 8];
      o[nt] = wmma_bf(ap.v, bv.v, o[nt]);
    }
  }
#pragma unroll
  for (int nt = 0; nt < 4; ++nt)
#pragma unroll
    for (int v = 0; v < 8; ++v) {
      int l = wid * 16 + v + 8 * hl;
      int d = nt * 16 + r;
      O[((long)n * 128 + l) * 768 + h * 64 + d] = o[nt][v];
    }
}

// ------------------------- residual + LayerNorm ----------------------------
__global__ __launch_bounds__(256) void addln_k(float* __restrict__ out,
                                               const float* __restrict__ x,
                                               const float* __restrict__ h,
                                               const float* __restrict__ g,
                                               const float* __restrict__ b) {
  __shared__ float red[256];
  const long base = (long)blockIdx.x * 768;
  const int tid = threadIdx.x;
  float v[3];
#pragma unroll
  for (int i = 0; i < 3; ++i) { int d = tid + i * 256; v[i] = x[base + d] + h[base + d]; }
  float s = v[0] + v[1] + v[2];
  red[tid] = s; __syncthreads();
  for (int o = 128; o > 0; o >>= 1) { if (tid < o) red[tid] += red[tid + o]; __syncthreads(); }
  float mean = red[0] * (1.f / 768.f);
  __syncthreads();
  float s2 = 0.f;
#pragma unroll
  for (int i = 0; i < 3; ++i) { float dv = v[i] - mean; s2 += dv * dv; }
  red[tid] = s2; __syncthreads();
  for (int o = 128; o > 0; o >>= 1) { if (tid < o) red[tid] += red[tid + o]; __syncthreads(); }
  float rstd = rsqrtf(red[0] * (1.f / 768.f) + 1e-5f);
#pragma unroll
  for (int i = 0; i < 3; ++i) {
    int d = tid + i * 256;
    out[base + d] = (v[i] - mean) * rstd * g[d] + b[d];
  }
}

// ------------------------------ embeddings ---------------------------------
__global__ __launch_bounds__(256) void embed_k(const int* __restrict__ ids,
                                               const int* __restrict__ spk,
                                               const float* __restrict__ tok,
                                               const float* __restrict__ pos,
                                               const float* __restrict__ spkE,
                                               float* __restrict__ x) {
  long t = blockIdx.x;                               // flat [B,T,L]
  int l = (int)(t & 127);
  long bt = t >> 7;
  long id = ids[t]; int s = spk[bt];
  const float* te = tok + id * 768;
  const float* pe = pos + (long)l * 768;
  const float* se = spkE + (long)s * 768;
#pragma unroll
  for (int i = 0; i < 3; ++i) {
    int d = threadIdx.x + i * 256;
    x[t * 768 + d] = te[d] + pe[d] + se[d];
  }
}

// --------------------------- masked mean pool ------------------------------
__global__ __launch_bounds__(256) void pool_k(const float* __restrict__ x,
                                              const int* __restrict__ mask,
                                              float* __restrict__ edus) {
  long bt = blockIdx.x;
  const int* am = mask + bt * 128;
  float cnt = 0.f;
  for (int l = 0; l < 128; ++l) cnt += (am[l] > 0) ? 1.f : 0.f;
  float inv = 1.f / fmaxf(cnt, 1e-6f);
#pragma unroll
  for (int i = 0; i < 3; ++i) {
    int d = threadIdx.x + i * 256;
    float acc = 0.f;
    for (int l = 0; l < 128; ++l)
      if (am[l] > 0) acc += x[(bt * 128 + l) * 768 + d];
    edus[bt * 768 + d] = acc * inv;
  }
}

// --------------------------- retrieval scores ------------------------------
__global__ void scores_k(const float* __restrict__ sq, const float* __restrict__ cq,
                         const float* __restrict__ sp, const float* __restrict__ cp,
                         float* __restrict__ sc) {
  int bt = blockIdx.x, b = bt >> 6, t = bt & 63, s = threadIdx.x;  // 64 threads
  float out;
  if (s < t) {
    const float* a1 = sq + (long)bt * 768;
    const float* a2 = cq + (long)bt * 768;
    const float* p1 = sp + (long)(b * 64 + s) * 768;
    const float* p2 = cp + (long)(b * 64 + s) * 768;
    float acc = 0.f;
    for (int k = 0; k < 768; ++k) acc += a1[k] * p1[k] + a2[k] * p2[k];
    out = acc;
  } else out = -__builtin_inff();
  sc[(long)bt * 64 + s] = out;
}

// --------------------------- top-K select + avg ----------------------------
__global__ __launch_bounds__(256) void topk_k(const float* __restrict__ sc,
                                              const float* __restrict__ edus,
                                              float* __restrict__ summ) {
  __shared__ int sidx[5];
  __shared__ int scnt;
  int bt = blockIdx.x, b = bt >> 6, t = bt & 63;
  if (threadIdx.x == 0) {
    float loc[64];
    for (int s = 0; s < 64; ++s) loc[s] = sc[(long)bt * 64 + s];
    int cnt = 0;
    for (int j = 0; j < 5; ++j) {
      float best = -__builtin_inff(); int bi = -1;
      for (int s = 0; s < t; ++s) if (loc[s] > best) { best = loc[s]; bi = s; }
      if (bi >= 0 && best > -1e37f) { sidx[cnt++] = bi; loc[bi] = -__builtin_inff(); }
    }
    scnt = cnt;
  }
  __syncthreads();
  int cnt = scnt;
  float inv = 1.f / fmaxf((float)cnt, 1.f);
#pragma unroll
  for (int i = 0; i < 3; ++i) {
    int d = threadIdx.x + i * 256;
    float acc = 0.f;
    for (int j = 0; j < cnt; ++j) acc += edus[(long)(b * 64 + sidx[j]) * 768 + d];
    summ[(long)bt * 768 + d] = acc * inv;
  }
}

// ------------------------------- GRU scan ----------------------------------
// One block per batch; mem[8][768] lives in LDS; gi precomputed by GEMM.
__global__ __launch_bounds__(1024) void gru_k(const float* __restrict__ gi,
                                              const float* __restrict__ Whh,
                                              const float* __restrict__ bhh,
                                              const int* __restrict__ spk,
                                              float* __restrict__ memOut) {
  __shared__ float mem_s[8 * 768];
  __shared__ float h_s[768];
  __shared__ float gh_s[2304];
  const int b = blockIdx.x, tid = threadIdx.x;
  for (int i = tid; i < 6144; i += 1024) mem_s[i] = 0.f;
  __syncthreads();
  for (int t = 1; t < 64; ++t) {
    int s = spk[b * 64 + t];
    for (int d = tid; d < 768; d += 1024) h_s[d] = mem_s[s * 768 + d];
    __syncthreads();
    for (int row = tid; row < 2304; row += 1024) {
      const float* w = Whh + (long)row * 768;
      float acc = bhh[row];
      for (int k = 0; k < 768; ++k) acc += w[k] * h_s[k];
      gh_s[row] = acc;
    }
    __syncthreads();
    const float* g = gi + (long)(b * 64 + t) * 2304;
    for (int d = tid; d < 768; d += 1024) {
      float r = sigmoidf_(g[d] + gh_s[d]);
      float z = sigmoidf_(g[768 + d] + gh_s[768 + d]);
      float nn = tanhf(g[1536 + d] + r * gh_s[1536 + d]);
      mem_s[s * 768 + d] = (1.f - z) * nn + z * h_s[d];
    }
    __syncthreads();
  }
  for (int i = tid; i < 6144; i += 1024) memOut[(long)b * 6144 + i] = mem_s[i];
}

// ------------------------- broadcast memory add ----------------------------
__global__ __launch_bounds__(256) void addmem_k(float* __restrict__ x,
                                                const float* __restrict__ memG,
                                                const int* __restrict__ spk) {
  long tkn = blockIdx.x;
  long bt = tkn >> 7, b = bt >> 6;
  int s = spk[bt];
  const float* m = memG + (b * 8 + s) * 768;
#pragma unroll
  for (int i = 0; i < 3; ++i) {
    int d = threadIdx.x + i * 256;
    x[tkn * 768 + d] += m[d];
  }
}

// ===========================================================================
extern "C" void kernel_launch(void* const* d_in, const int* in_sizes, int n_in,
                              void* d_out, int out_size, void* d_ws, size_t ws_size,
                              hipStream_t stream) {
  (void)in_sizes; (void)n_in; (void)out_size; (void)ws_size;
  // d_in layout: [0]=input_ids [1]=attention_mask [2]=speaker_ids, then params
  // flattened in jax pytree order (dict keys sorted, lists in order):
  //   Wcp Wcq Wsp Wsq bcp bcq bsp bsq | final[0] final[1] |
  //   gru_Whh gru_Wih gru_bhh gru_bih | init[0] init[1] | pos_emb spk_emb tok_emb
  // per-layer order: W1 W2 Wk Wo Wq Wv b1 b2 bk bo bq bv ln1_b ln1_g ln2_b ln2_g
  const int PB = 3;
  const int P_WCP = PB + 0, P_WCQ = PB + 1, P_WSP = PB + 2, P_WSQ = PB + 3;
  const int P_BCP = PB + 4, P_BCQ = PB + 5, P_BSP = PB + 6, P_BSQ = PB + 7;
  const int L_FIN0 = PB + 8, L_FIN1 = PB + 24;
  const int P_GRU_WHH = PB + 40, P_GRU_WIH = PB + 41;
  const int P_GRU_BHH = PB + 42, P_GRU_BIH = PB + 43;
  const int L_INIT0 = PB + 44, L_INIT1 = PB + 60;
  const int P_POS = PB + 76, P_SPKE = PB + 77, P_TOK = PB + 78;
  // per-layer offsets
  const int LO_W1 = 0, LO_W2 = 1, LO_WK = 2, LO_WO = 3, LO_WQ = 4, LO_WV = 5;
  const int LO_B1 = 6, LO_B2 = 7, LO_BK = 8, LO_BO = 9, LO_BQ = 10, LO_BV = 11;
  const int LO_LN1B = 12, LO_LN1G = 13, LO_LN2B = 14, LO_LN2G = 15;

  const int* ids  = (const int*)d_in[0];
  const int* am   = (const int*)d_in[1];
  const int* spk  = (const int*)d_in[2];

  const long MTOK = 32768;                     // B*T*L tokens
  char* ws = (char*)d_ws;
  size_t off = 0;
  auto carve = [&](size_t bytes) { size_t p = off; off += (bytes + 255) & ~(size_t)255; return p; };
  const size_t oXB  = carve(MTOK * 768 * 4);   // main f32 activations
  const size_t oT1  = carve(MTOK * 3072 * 2);  // bf16 arena (qkv / ffn hidden / o f32)
  const size_t oT2  = carve(MTOK * 768 * 4);   // f32 scratch (attn out, ffn out)
  const size_t oWB  = carve((size_t)32440320 * 2); // bf16 weights
  const size_t oEDU = carve(256 * 768 * 4);
  const size_t oSQ  = carve(256 * 768 * 4);
  const size_t oCQ  = carve(256 * 768 * 4);
  const size_t oSP  = carve(256 * 768 * 4);
  const size_t oCP  = carve(256 * 768 * 4);
  const size_t oSC  = carve(4 * 64 * 64 * 4);
  const size_t oSUM = carve(256 * 768 * 4);
  const size_t oGI  = carve(256 * 2304 * 4);
  const size_t oMEM = carve(4 * 8 * 768 * 4);

  float* XB   = (float*)(ws + oXB);
  float* T2f  = (float*)(ws + oT2);
  unsigned short* qB = (unsigned short*)(ws + oT1);
  unsigned short* kB = qB + MTOK * 768;
  unsigned short* vB = kB + MTOK * 768;
  float* oF   = (float*)(ws + oT1);            // overlaps dead q/k region
  unsigned short* hB = (unsigned short*)(ws + oT1);
  float* edus = (float*)(ws + oEDU);
  float* sqF  = (float*)(ws + oSQ);
  float* cqF  = (float*)(ws + oCQ);
  float* spF  = (float*)(ws + oSP);
  float* cpF  = (float*)(ws + oCP);
  float* scF  = (float*)(ws + oSC);
  float* sumF = (float*)(ws + oSUM);
  float* giF  = (float*)(ws + oGI);
  float* memF = (float*)(ws + oMEM);

  // ---- convert weights to bf16 ----
  unsigned short* WBp = (unsigned short*)(ws + oWB);
  size_t woff = 0;
  auto wal = [&](size_t n) { unsigned short* p = WBp + woff; woff += n; return p; };
  auto cvt = [&](const void* src, unsigned short* dst, long n) {
    long n4 = n / 4;
    cvt_k<<<(unsigned)((n4 + 255) / 256), 256, 0, stream>>>((const float*)src, dst, n4);
  };
  struct LayerW { const unsigned short *Wq, *Wk, *Wv, *Wo, *W1, *W2; };
  LayerW LW[4];
  const int bases[4] = {L_INIT0, L_INIT1, L_FIN0, L_FIN1};
  for (int i = 0; i < 4; ++i) {
    int b = bases[i];
    unsigned short* p;
    p = wal(589824);  cvt(d_in[b + LO_WQ], p, 589824);  LW[i].Wq = p;
    p = wal(589824);  cvt(d_in[b + LO_WK], p, 589824);  LW[i].Wk = p;
    p = wal(589824);  cvt(d_in[b + LO_WV], p, 589824);  LW[i].Wv = p;
    p = wal(589824);  cvt(d_in[b + LO_WO], p, 589824);  LW[i].Wo = p;
    p = wal(2359296); cvt(d_in[b + LO_W1], p, 2359296); LW[i].W1 = p;
    p = wal(2359296); cvt(d_in[b + LO_W2], p, 2359296); LW[i].W2 = p;
  }
  unsigned short *WsqB = wal(589824), *WcqB = wal(589824);
  unsigned short *WspB = wal(589824), *WcpB = wal(589824);
  unsigned short *WihB = wal(1769472);
  cvt(d_in[P_WSQ], WsqB, 589824); cvt(d_in[P_WCQ], WcqB, 589824);
  cvt(d_in[P_WSP], WspB, 589824); cvt(d_in[P_WCP], WcpB, 589824);
  cvt(d_in[P_GRU_WIH], WihB, 1769472);

  auto gemm = [&](const void* A, const unsigned short* W, const float* bias, void* C,
                  int M, int N, int Kd, int act, int abf, int obf) {
    dim3 g(N / 128, M / 128), blk(256);
    if (!act && !abf && !obf)      gemm_k<0,0,0><<<g, blk, 0, stream>>>(A, W, bias, C, M, N, Kd);
    else if (!act && !abf && obf)  gemm_k<0,0,1><<<g, blk, 0, stream>>>(A, W, bias, C, M, N, Kd);
    else if (act && !abf && obf)   gemm_k<1,0,1><<<g, blk, 0, stream>>>(A, W, bias, C, M, N, Kd);
    else                           gemm_k<0,1,0><<<g, blk, 0, stream>>>(A, W, bias, C, M, N, Kd);
  };
  auto F32 = [&](int i) { return (const float*)d_in[i]; };

  auto encLayer = [&](int b, const LayerW& lw, float* lnOut) {
    gemm(XB, lw.Wq, F32(b + LO_BQ), qB, 32768, 768, 768, 0, 0, 1);
    gemm(XB, lw.Wk, F32(b + LO_BK), kB, 32768, 768, 768, 0, 0, 1);
    gemm(XB, lw.Wv, F32(b + LO_BV), vB, 32768, 768, 768, 0, 0, 1);
    attn_k<<<256 * 12, 256, 0, stream>>>(qB, kB, vB, am, T2f, 12);
    gemm(T2f, lw.Wo, F32(b + LO_BO), oF, 32768, 768, 768, 0, 0, 0);
    addln_k<<<32768, 256, 0, stream>>>(XB, XB, oF, F32(b + LO_LN1G), F32(b + LO_LN1B));
    gemm(XB, lw.W1, F32(b + LO_B1), hB, 32768, 3072, 768, 1, 0, 1);   // GELU -> bf16
    gemm(hB, lw.W2, F32(b + LO_B2), T2f, 32768, 768, 3072, 0, 1, 0);
    addln_k<<<32768, 256, 0, stream>>>(lnOut, XB, T2f, F32(b + LO_LN2G), F32(b + LO_LN2B));
  };

  // ---- pipeline ----
  embed_k<<<32768, 256, 0, stream>>>(ids, spk, F32(P_TOK), F32(P_POS), F32(P_SPKE), XB);
  encLayer(L_INIT0, LW[0], XB);
  encLayer(L_INIT1, LW[1], XB);
  pool_k<<<256, 256, 0, stream>>>(XB, am, edus);
  gemm(edus, WsqB, F32(P_BSQ), sqF, 256, 768, 768, 0, 0, 0);
  gemm(edus, WcqB, F32(P_BCQ), cqF, 256, 768, 768, 0, 0, 0);
  gemm(edus, WspB, F32(P_BSP), spF, 256, 768, 768, 0, 0, 0);
  gemm(edus, WcpB, F32(P_BCP), cpF, 256, 768, 768, 0, 0, 0);
  scores_k<<<256, 64, 0, stream>>>(sqF, cqF, spF, cpF, scF);
  topk_k<<<256, 256, 0, stream>>>(scF, edus, sumF);
  gemm(sumF, WihB, F32(P_GRU_BIH), giF, 256, 2304, 768, 0, 0, 0);
  gru_k<<<4, 1024, 0, stream>>>(giF, F32(P_GRU_WHH), F32(P_GRU_BHH), spk, memF);
  addmem_k<<<32768, 256, 0, stream>>>(XB, memF, spk);
  encLayer(L_FIN0, LW[2], XB);
  encLayer(L_FIN1, LW[3], (float*)d_out);
}
